// DeepSeekMoERouter_35871566856547
// MI455X (gfx1250) — matrix operand order repaired and checked by reading, not compiled, not run on hardware
//
#include <hip/hip_runtime.h>
#include <hip/hip_bf16.h>

// ---- problem constants (match reference) ----
#define BATCH   4
#define SEQ     4096
#define TOKENS  (BATCH * SEQ)   // 16384
#define DDIM    2048
#define NGRP    4
#define EPGC    16
#define NEXP    64               // NGRP * EPGC
#define HDIM    64
#define NCOLS   144              // 64 (W1) + 4 (Wg) + 64 (We) padded to 9*16
#define NT      9                // N tiles of 16
#define KCH     32               // K chunk (WMMA K)
#define NKCH    (DDIM / KCH)     // 64

typedef __attribute__((ext_vector_type(16))) __bf16        v16bf;
typedef __attribute__((ext_vector_type(16))) unsigned short v16u;
typedef __attribute__((ext_vector_type(8)))  float          v8f;
typedef __attribute__((ext_vector_type(4)))  unsigned int   u32x4;
typedef __attribute__((ext_vector_type(8)))  int            i32x8;
typedef __attribute__((ext_vector_type(4)))  int            i32x4;

__device__ __forceinline__ unsigned short f2bf(float f) {
    // round-to-nearest-even f32 -> bf16
    unsigned u = __builtin_bit_cast(unsigned, f);
    return (unsigned short)((u + 0x7FFFu + ((u >> 16) & 1u)) >> 16);
}

// ---------------------------------------------------------------------------
// Kernel 1: pack weights [144][2048] bf16 row-major (n,k); zero aux accums.
// Row n: 0..63 -> W1[h], 64..67 -> Wg[g], 68..131 -> We[g*16+e], 132..143 -> 0
// ---------------------------------------------------------------------------
__global__ __launch_bounds__(256) void moe_prep_kernel(
    const float* __restrict__ W1, const float* __restrict__ Wg,
    const float* __restrict__ We, unsigned short* __restrict__ wb,
    float* __restrict__ aux_acc)
{
    int i = blockIdx.x * 256 + threadIdx.x;
    if (i < 128) aux_acc[i] = 0.0f;          // [0..63]=router sums, [64..127]=dispatch sums
    if (i >= NCOLS * DDIM) return;
    int n = i / DDIM;
    int k = i - n * DDIM;
    float v = 0.0f;
    if      (n < 64)  v = W1[n * DDIM + k];
    else if (n < 68)  v = Wg[(n - 64) * DDIM + k];
    else if (n < 132) v = We[(n - 68) * DDIM + k];
    wb[i] = f2bf(v);
}

// ---------------------------------------------------------------------------
// TDM: DMA one K-chunk of the packed weights (144 rows x 32 bf16 = 9 KB)
// into LDS. D# per CDNA5 ISA:
//   group0: count=1 | lds_addr | global_addr(57b) | type=2
//   group1: data_size=2B, tensor_dim0=2048, tensor_dim1=144,
//           tile_dim0=32, tile_dim1=144, tensor_dim0_stride=2048
//   groups 2/3 (+ extra zero group on this toolchain): zero (2D tensor)
// ---------------------------------------------------------------------------
__device__ __forceinline__ void tdm_load_chunk(const unsigned short* wb, int kb,
                                               unsigned lds_off)
{
    unsigned long long ga =
        (unsigned long long)(const void*)wb + (unsigned long long)(kb * 2);
    u32x4 g0;
    g0[0] = 1u;                                            // count=1 (valid user D#)
    g0[1] = lds_off;                                       // lds_addr (bytes)
    g0[2] = (unsigned)(ga & 0xFFFFFFFFu);                  // global_addr[31:0]
    g0[3] = (unsigned)((ga >> 32) & 0x1FFFFFFu)            // global_addr[56:32]
          | (2u << 30);                                    // type=2 ("image")
    i32x8 g1;
    g1[0] = 0x00010000;                                    // data_size=1 -> 2 bytes
    g1[1] = (int)(2048u << 16);                            // tensor_dim0[15:0]=2048
    g1[2] = (int)(144u << 16);                             // dim0 hi=0, tensor_dim1=144
    g1[3] = (int)(32u << 16);                              // dim1 hi=0, tile_dim0=32
    g1[4] = 144;                                           // tile_dim1=144, tile_dim2=0
    g1[5] = 2048;                                          // tensor_dim0_stride lo
    g1[6] = 0;                                             // stride hi / dim1_stride lo
    g1[7] = 0;
    i32x4 z4 = (i32x4){0, 0, 0, 0};
    i32x8 z8 = (i32x8){0, 0, 0, 0, 0, 0, 0, 0};
    __builtin_amdgcn_tensor_load_to_lds(g0, g1, z4, z4, z8, 0);
}

// ---------------------------------------------------------------------------
// Kernel 2: WMMA GEMM  logits[t][n] = sum_k x[t][k] * wb[n][k]
// 256 threads = 8 waves; each wave owns 16 tokens x all 9 N-tiles.
// B tiles double-buffered in LDS via TDM; A streamed from global (f32->bf16).
// ---------------------------------------------------------------------------
__global__ __launch_bounds__(256) void moe_gemm_kernel(
    const float* __restrict__ x, const unsigned short* __restrict__ wb,
    float* __restrict__ logits)
{
    __shared__ __align__(64) unsigned short s_b[2][NCOLS][KCH];   // 2 x 9 KB

    const int wave = threadIdx.x >> 5;
    const int lane = threadIdx.x & 31;
    const int hi   = lane >> 4;            // 0: low-K half lanes, 1: high-K half
    const int ln   = lane & 15;            // row (A/C) or column (B) within tile
    const int tokenBase = (blockIdx.x * 8 + wave) * 16;

    const unsigned lds0 =
        (unsigned)(unsigned long long)(const void*)&s_b[0][0][0];
    const unsigned lds1 =
        (unsigned)(unsigned long long)(const void*)&s_b[1][0][0];

    v8f acc[NT];
#pragma unroll
    for (int i = 0; i < NT; ++i) acc[i] = (v8f){0, 0, 0, 0, 0, 0, 0, 0};

    const float* xrow = x + (size_t)(tokenBase + ln) * DDIM;
    const int klo = hi * 16;               // A-frag: lanes>=16 take K+16..31

    if (wave == 0) tdm_load_chunk(wb, 0, lds0);     // prologue: fill buffer 0

    for (int kbi = 0; kbi < NKCH; ++kbi) {
        const int kb = kbi * KCH;

        if (wave == 0) __builtin_amdgcn_s_wait_tensorcnt(0);  // my TDM landed
        __syncthreads();                                       // visible to all waves

        if (wave == 0 && (kbi + 1) < NKCH)                     // prefetch next chunk
            tdm_load_chunk(wb, kb + KCH, (kbi & 1) ? lds0 : lds1);

        // A fragment: row ln, contiguous K (element s <- k = 16*hi + s), same order as B
        v16u au;
#pragma unroll
        for (int j = 0; j < 8; ++j) au[j]     = f2bf(xrow[kb + klo + j]);
#pragma unroll
        for (int j = 0; j < 8; ++j) au[8 + j] = f2bf(xrow[kb + klo + 8 + j]);
        v16bf a = __builtin_bit_cast(v16bf, au);

        const unsigned short* bbase = &s_b[kbi & 1][0][0];
#pragma unroll
        for (int nt = 0; nt < NT; ++nt) {
            // B fragment from LDS: column n = nt*16+ln; lanes 0-15 K 0..15,
            // lanes 16-31 K 16..31 (32B contiguous, 32B aligned -> ds_load_b128 x2)
            v16bf b = *(const v16bf*)(bbase + (nt * 16 + ln) * KCH + hi * 16);
            acc[nt] = __builtin_amdgcn_wmma_f32_16x16x32_bf16(
                false, a, false, b, (short)0, acc[nt], false, false);
        }
        // single barrier/iter is enough: __syncthreads at top of next iteration
        // drains DScnt for all waves before the buffer is overwritten.
    }

    // C layout: VGPR r -> M = r + 8*hi (lanes 16-31), N = ln
#pragma unroll
    for (int nt = 0; nt < NT; ++nt) {
#pragma unroll
        for (int r = 0; r < 8; ++r) {
            int m = tokenBase + r + hi * 8;
            logits[(size_t)m * NCOLS + nt * 16 + ln] = acc[nt][r];
        }
    }
}

// ---------------------------------------------------------------------------
// Kernel 3: per-token epilogue + block-level reduction for aux loss.
// ---------------------------------------------------------------------------
__global__ __launch_bounds__(256) void moe_epilogue_kernel(
    const float* __restrict__ logits,
    const float* __restrict__ b1, const float* __restrict__ W2,
    const float* __restrict__ b2,
    const int* __restrict__ min_e_p, const int* __restrict__ max_e_p,
    float* __restrict__ out, float* __restrict__ aux_acc)
{
    __shared__ float s_r[NEXP];
    __shared__ float s_d[NEXP];
    for (int i = threadIdx.x; i < NEXP; i += 256) { s_r[i] = 0.0f; s_d[i] = 0.0f; }
    __syncthreads();

    const int t = blockIdx.x * 256 + threadIdx.x;   // one thread per token
    const float* row = logits + (size_t)t * NCOLS;
    const int min_e = *min_e_p;
    const int max_e = *max_e_p;

    // complexity: sigmoid(W2 . relu(W1x + b1) + b2)
    float c = b2[0];
#pragma unroll 8
    for (int h = 0; h < HDIM; ++h) {
        float hv = row[h] + b1[h];
        hv = hv > 0.0f ? hv : 0.0f;
        c += hv * W2[h];
    }
    float comp = 1.0f / (1.0f + expf(-c));
    int k = (int)(comp * (float)max_e);             // trunc toward zero (comp >= 0)
    k = k < min_e ? min_e : (k > max_e ? max_e : k);

    // group softmax / argmax (argmax of logits == argmax of probs)
    float gl[NGRP];
#pragma unroll
    for (int g = 0; g < NGRP; ++g) gl[g] = row[64 + g];
    int   gidx = 0;
    float gmax = gl[0];
#pragma unroll
    for (int g = 1; g < NGRP; ++g) if (gl[g] > gmax) { gmax = gl[g]; gidx = g; }
    float gsum = 0.0f;
#pragma unroll
    for (int g = 0; g < NGRP; ++g) gsum += expf(gl[g] - gmax);
    const float gp = 1.0f / gsum;                   // group prob at argmax

    // chosen-group expert softmax
    float ep[EPGC];
    const float* erow = row + 68 + gidx * EPGC;
    float emax = erow[0];
#pragma unroll
    for (int e = 1; e < EPGC; ++e) emax = erow[e] > emax ? erow[e] : emax;
    float esum = 0.0f;
#pragma unroll
    for (int e = 0; e < EPGC; ++e) { ep[e] = expf(erow[e] - emax); esum += ep[e]; }
    const float einv = 1.0f / esum;
#pragma unroll
    for (int e = 0; e < EPGC; ++e) ep[e] *= einv;

    // top-2 (ties -> lowest index, matches lax.top_k)
    int i1 = 0; float v1 = ep[0];
#pragma unroll
    for (int e = 1; e < EPGC; ++e) if (ep[e] > v1) { v1 = ep[e]; i1 = e; }
    int i2 = (i1 == 0) ? 1 : 0; float v2 = ep[i2];
#pragma unroll
    for (int e = 0; e < EPGC; ++e)
        if (e != i1 && ep[e] > v2) { v2 = ep[e]; i2 = e; }

    const float use2  = (k >= 2) ? 1.0f : 0.0f;
    const float denom = v1 + use2 * v2;
    const float np1   = v1 / denom;
    const float np2   = use2 * v2 / denom;

    float* drow = out + (size_t)t * NEXP;
    float* crow = out + (size_t)TOKENS * NEXP + (size_t)t * NEXP;
    float* prow = out + 2 * (size_t)TOKENS * NEXP + (size_t)t * NEXP;
#pragma unroll 16
    for (int e = 0; e < NEXP; ++e) {
        drow[e] = 0.0f;
        crow[e] = 0.0f;
        prow[e] = ((e >> 4) == gidx) ? gp * ep[e & 15] : 0.0f;
    }
    drow[gidx * EPGC + i1] = 1.0f;
    drow[gidx * EPGC + i2] = use2;
    crow[gidx * EPGC + i1] = np1;
    crow[gidx * EPGC + i2] = np2;    // i1 != i2 always

    // LDS reduction for aux loss sums (ds_add_f32)
    atomicAdd(&s_d[gidx * EPGC + i1], 1.0f);
    atomicAdd(&s_d[gidx * EPGC + i2], use2);
#pragma unroll
    for (int e = 0; e < EPGC; ++e) atomicAdd(&s_r[gidx * EPGC + e], gp * ep[e]);
    __syncthreads();

    for (int i = threadIdx.x; i < NEXP; i += 256) {
        atomicAdd(&aux_acc[i],        s_r[i]);
        atomicAdd(&aux_acc[NEXP + i], s_d[i]);
    }
}

// ---------------------------------------------------------------------------
// Kernel 4: aux_loss = sum_e (rsum/T)*(dsum/T) * E
// ---------------------------------------------------------------------------
__global__ __launch_bounds__(64) void moe_aux_kernel(
    const float* __restrict__ aux_acc, float* __restrict__ out_aux)
{
    __shared__ float s[NEXP];
    const float invT = 1.0f / (float)TOKENS;
    int e = threadIdx.x;
    s[e] = (aux_acc[e] * invT) * (aux_acc[NEXP + e] * invT);
    __syncthreads();
    if (e == 0) {
        float a = 0.0f;
#pragma unroll
        for (int i = 0; i < NEXP; ++i) a += s[i];
        out_aux[0] = a * (float)NEXP;
    }
}

// ---------------------------------------------------------------------------
extern "C" void kernel_launch(void* const* d_in, const int* in_sizes, int n_in,
                              void* d_out, int out_size, void* d_ws, size_t ws_size,
                              hipStream_t stream) {
    const float* x    = (const float*)d_in[0];
    const float* Wg   = (const float*)d_in[1];
    const float* We   = (const float*)d_in[2];
    const float* W1   = (const float*)d_in[3];
    const float* b1   = (const float*)d_in[4];
    const float* W2   = (const float*)d_in[5];
    const float* b2   = (const float*)d_in[6];
    const int* min_e  = (const int*)d_in[7];
    const int* max_e  = (const int*)d_in[8];
    float* out = (float*)d_out;

    // workspace layout
    char* ws = (char*)d_ws;
    unsigned short* wb = (unsigned short*)ws;                       // 144*2048*2  = 576 KB
    size_t off_logits  = (size_t)NCOLS * DDIM * sizeof(unsigned short);
    float* logits      = (float*)(ws + off_logits);                 // 16384*144*4 = 9.0 MB
    size_t off_aux     = off_logits + (size_t)TOKENS * NCOLS * sizeof(float);
    float* aux_acc     = (float*)(ws + off_aux);                    // 128 floats

    moe_prep_kernel<<<(NCOLS * DDIM + 255) / 256, 256, 0, stream>>>(
        W1, Wg, We, wb, aux_acc);

    moe_gemm_kernel<<<TOKENS / 128, 256, 0, stream>>>(x, wb, logits);

    moe_epilogue_kernel<<<TOKENS / 256, 256, 0, stream>>>(
        logits, b1, W2, b2, min_e, max_e, out, aux_acc);

    moe_aux_kernel<<<1, 64, 0, stream>>>(
        aux_acc, out + 3 * (size_t)TOKENS * NEXP);
}
